// density_wfa_finetune_46849503264859
// MI455X (gfx1250) — compile-verified, matrix operand order checked by reading, not run on hardware
//
#include <hip/hip_runtime.h>
#include <hip/hip_bf16.h>
#include <math.h>

typedef __bf16 bf16;
typedef __attribute__((ext_vector_type(16))) __bf16 v16bf;
typedef __attribute__((ext_vector_type(8)))  float  v8f;

#define WMMA_BF16(a, b, c) \
  __builtin_amdgcn_wmma_f32_16x16x32_bf16(false, (a), false, (b), (short)0, (c), false, false)

namespace {
constexpr int   NSAMP  = 8192;
constexpr int   TSTEPS = 128;
constexpr int   RDIM   = 128;   // r
constexpr int   EH     = 128;   // encoder hidden
constexpr int   DENC   = 64;    // d_enc
constexpr int   NH     = 256;   // nade hidden
constexpr float LOG2PI = 1.8378770664093453f;

constexpr int MT = 32;     // samples per workgroup
constexpr int S1 = 136;    // LDS stride (halves) for 128-wide buffers
constexpr int S2 = 264;    // LDS stride (halves) for 256-wide buffers
constexpr int SH = 100;    // heads f32 stride (mu|sig|alpha = 96 cols)
constexpr int SE = 72;     // enc stride (64 cols)
}

// ---------------------------------------------------------------------------
// WMMA operand loaders.
// A-operand (16x32 bf16), per CDNA5 ISA layout: lane holds row M = lane&15;
// lanes 0-15 carry K {0..7, 16..23}, lanes 16-31 carry K {8..15, 24..31}.
// Elements 0..7 -> K = kb..kb+7, elements 8..15 -> K = kb+16..kb+23.
__device__ __forceinline__ v16bf load_a_lds(const bf16* base, int stride, int lane, int k0) {
  const int m  = lane & 15;
  const int kb = k0 + ((lane & 16) >> 1);
  const bf16* p = base + m * stride + kb;
  v16bf a;
#pragma unroll
  for (int e = 0; e < 8; ++e) { a[e] = p[e]; a[e + 8] = p[16 + e]; }
  return a;
}

// B-operand: pre-packed so each lane's 16 halves are contiguous (32B/lane).
__device__ __forceinline__ v16bf load_b_pack(const bf16* block, int lane) {
  return *(const v16bf*)(block + lane * 16);
}

// ---------------------------------------------------------------------------
// Pack a matrix into WMMA B-operand blocks of 512 bf16 (32 lanes x 16 halves).
// Block (kc, ot) at offset (kc*(O/16)+ot)*512.
// isOK=1: W stored [O][K] row-major (weight, B[k][o] = W[o][k]).
// isOK=0: W stored [K][O] row-major (B[k][o] = W[k*O+o]).
__global__ void pack_b_kernel(const float* __restrict__ W, bf16* __restrict__ out,
                              int K, int O, int isOK) {
  int idx = blockIdx.x * blockDim.x + threadIdx.x;
  int total = K * O;
  if (idx >= total) return;
  int block  = idx >> 9;
  int within = idx & 511;
  int lane = within >> 4;
  int e    = within & 15;
  int otc = O >> 4;
  int kc = block / otc;
  int ot = block - kc * otc;
  int n  = (ot << 4) + (lane & 15);
  int kb = (kc << 5) + ((lane & 16) >> 1);
  int k  = kb + (e < 8 ? e : e + 8);
  float v = isOK ? W[n * K + k] : W[k * O + n];
  out[idx] = (bf16)v;
}

// ---------------------------------------------------------------------------
__global__ void init_kernel(const float* __restrict__ init_w,
                            float* __restrict__ tmp0,
                            float* __restrict__ out,
                            float* __restrict__ nrmsq) {
  int idx = blockIdx.x * blockDim.x + threadIdx.x;
  if (idx < NSAMP * RDIM) tmp0[idx] = init_w[idx & (RDIM - 1)];
  if (idx <= NSAMP)  out[idx]   = 0.f;
  if (idx < TSTEPS)  nrmsq[idx] = 0.f;
}

__global__ void final_kernel(const float* __restrict__ init_w,
                             const float* __restrict__ nrmsq,
                             float* __restrict__ out) {
  __shared__ float sh[TSTEPS];
  int t = threadIdx.x;
  sh[t] = (t >= 1) ? sqrtf(nrmsq[t]) : 0.f;
  __syncthreads();
  if (t == 0) {
    float s = 0.f;
    for (int i = 0; i < TSTEPS; ++i) s += sh[i];
    float iw = 0.f;
    for (int i = 0; i < RDIM; ++i) { float w = init_w[i]; iw += w * w; }
    out[NSAMP] = s + sqrtf(iw);
  }
}

// ---------------------------------------------------------------------------
// One recurrence step, fully fused: phi (MLP + mixture loglik), ||tmp||_F
// partial, encoding, and nxt = einsum('nd,ni,idj->nj', enc, tmp, A).
__global__ __launch_bounds__(256) void step_kernel(
    const float* __restrict__ X,
    const float* __restrict__ e1w, const float* __restrict__ e1b,
    const float* __restrict__ e2b,
    const float* __restrict__ n1b, const float* __restrict__ n2b,
    const float* __restrict__ mub, const float* __restrict__ sgb,
    const float* __restrict__ alb,
    const bf16* __restrict__ P_n1, const bf16* __restrict__ P_n2,
    const bf16* __restrict__ P_mu, const bf16* __restrict__ P_sg,
    const bf16* __restrict__ P_al, const bf16* __restrict__ P_e2,
    const bf16* __restrict__ P_A,
    const float* __restrict__ tmp_in, float* __restrict__ tmp_out,
    float* __restrict__ res_out, float* __restrict__ nrmsq, int t) {
  const int tid  = threadIdx.x;
  const int lane = tid & 31;
  const int wave = tid >> 5;
  const int s0   = blockIdx.x * MT;

  __shared__ float xs[MT];
  __shared__ __align__(16) bf16 tmpb[MT * S1];   // raw tmp (bf16) for nxt
  __shared__ __align__(16) bf16 bufA[MT * S1];   // tanh(tmp), later e1
  __shared__ __align__(16) bf16 bufB[MT * S2];   // h1, later aliased as heads(f32)
  __shared__ __align__(16) bf16 bufC[MT * S2];   // h2
  __shared__ __align__(16) bf16 encb[MT * SE];   // encoding (bf16)
  __shared__ float redsum;

  if (tid == 0) redsum = 0.f;
  if (tid < MT) xs[tid] = X[(s0 + tid) * TSTEPS + t];

  // Phase 0: load tmp tile, sum of squares, tanh -> bufA, raw -> tmpb.
  float ss = 0.f;
  for (int idx = tid; idx < MT * RDIM; idx += 256) {
    int m = idx >> 7, i = idx & 127;
    float v = tmp_in[(s0 + m) * RDIM + i];
    ss += v * v;
    tmpb[m * S1 + i] = (bf16)v;
    bufA[m * S1 + i] = (bf16)tanhf(v);
  }
#pragma unroll
  for (int off = 16; off > 0; off >>= 1) ss += __shfl_xor(ss, off, 32);
  __syncthreads();                       // B1
  if (lane == 0) atomicAdd(&redsum, ss);

  // Phase 1: h1 = relu(tanh(tmp) @ nade1_w.T + b1)  [32x128]@[128x256]
#pragma unroll
  for (int q = 0; q < 4; ++q) {
    int tile = wave + (q << 3);
    int mt = tile >> 4, ot = tile & 15;
    v8f c = {};
#pragma unroll
    for (int kc = 0; kc < 4; ++kc) {
      v16bf a = load_a_lds(bufA + mt * 16 * S1, S1, lane, kc << 5);
      v16bf b = load_b_pack(P_n1 + ((kc << 4) + ot) * 512, lane);
      c = WMMA_BF16(a, b, c);
    }
    int col = (ot << 4) + (lane & 15);
    float bias = n1b[col];
    int rbase = (mt << 4) + ((lane & 16) >> 1);
#pragma unroll
    for (int e = 0; e < 8; ++e)
      bufB[(rbase + e) * S2 + col] = (bf16)fmaxf(c[e] + bias, 0.f);
  }
  __syncthreads();                       // B2
  if (tid == 0) atomicAdd(&nrmsq[t], redsum);

  // Phase 2: e1 = relu(x * enc1_w + enc1_b) -> bufA (h0 dead)
  for (int idx = tid; idx < MT * EH; idx += 256) {
    int m = idx >> 7, q2 = idx & 127;
    bufA[m * S1 + q2] = (bf16)fmaxf(xs[m] * e1w[q2] + e1b[q2], 0.f);
  }
  __syncthreads();                       // B3

  // Phase 3: h2 = relu(h1 @ nade2_w.T + b2)  [32x256]@[256x256]
#pragma unroll
  for (int q = 0; q < 4; ++q) {
    int tile = wave + (q << 3);
    int mt = tile >> 4, ot = tile & 15;
    v8f c = {};
#pragma unroll
    for (int kc = 0; kc < 8; ++kc) {
      v16bf a = load_a_lds(bufB + mt * 16 * S2, S2, lane, kc << 5);
      v16bf b = load_b_pack(P_n2 + ((kc << 4) + ot) * 512, lane);
      c = WMMA_BF16(a, b, c);
    }
    int col = (ot << 4) + (lane & 15);
    float bias = n2b[col];
    int rbase = (mt << 4) + ((lane & 16) >> 1);
#pragma unroll
    for (int e = 0; e < 8; ++e)
      bufC[(rbase + e) * S2 + col] = (bf16)fmaxf(c[e] + bias, 0.f);
  }
  __syncthreads();                       // B4 (bufB/h1 now dead)

  // Phase 4: enc = tanh(e1 @ enc2_w.T + b) and heads mu/sig/alpha from h2.
  float* headb = (float*)bufB;           // alias onto dead h1 buffer
  for (int j = wave; j < 20; j += 8) {
    if (j < 8) {                         // enc tiles: 2 mt x 4 ot
      int mt = j >> 2, ot = j & 3;
      v8f c = {};
#pragma unroll
      for (int kc = 0; kc < 4; ++kc) {
        v16bf a = load_a_lds(bufA + mt * 16 * S1, S1, lane, kc << 5);
        v16bf b = load_b_pack(P_e2 + ((kc << 2) + ot) * 512, lane);
        c = WMMA_BF16(a, b, c);
      }
      int col = (ot << 4) + (lane & 15);
      float bias = e2b[col];
      int rbase = (mt << 4) + ((lane & 16) >> 1);
#pragma unroll
      for (int e = 0; e < 8; ++e)
        encb[(rbase + e) * SE + col] = (bf16)tanhf(c[e] + bias);
    } else {                             // head tiles: 3 heads x 2 mt x 2 ot
      int hj = j - 8;
      int head = hj >> 2;
      int sub = hj & 3, mt = sub >> 1, ot = sub & 1;
      const bf16* P = head == 0 ? P_mu : (head == 1 ? P_sg : P_al);
      const float* bb = head == 0 ? mub : (head == 1 ? sgb : alb);
      v8f c = {};
#pragma unroll
      for (int kc = 0; kc < 8; ++kc) {
        v16bf a = load_a_lds(bufC + mt * 16 * S2, S2, lane, kc << 5);
        v16bf b = load_b_pack(P + ((kc << 1) + ot) * 512, lane);
        c = WMMA_BF16(a, b, c);
      }
      int col = (ot << 4) + (lane & 15);
      float bias = bb[col];
      int rbase = (mt << 4) + ((lane & 16) >> 1);
#pragma unroll
      for (int e = 0; e < 8; ++e)
        headb[(rbase + e) * SH + head * 32 + col] = c[e] + bias;
    }
  }
  __syncthreads();                       // B5

  // Phase 5: mixture log-likelihood (k=32 == 32 lanes, wave32 reduction).
  for (int q = 0; q < 4; ++q) {
    int m = wave * 4 + q;
    float x  = xs[m];
    float mu = headb[m * SH + lane];
    float ls = headb[m * SH + 32 + lane];
    float al = headb[m * SH + 64 + lane];
    float z = (x - mu) * expf(-ls);
    float comp = -0.5f * z * z - ls - 0.5f * LOG2PI;
    float a1 = al + comp;
    float m1 = a1, m2 = al;
#pragma unroll
    for (int off = 16; off > 0; off >>= 1) {
      m1 = fmaxf(m1, __shfl_xor(m1, off, 32));
      m2 = fmaxf(m2, __shfl_xor(m2, off, 32));
    }
    float s1 = expf(a1 - m1), s2 = expf(al - m2);
#pragma unroll
    for (int off = 16; off > 0; off >>= 1) {
      s1 += __shfl_xor(s1, off, 32);
      s2 += __shfl_xor(s2, off, 32);
    }
    if (lane == 0)
      res_out[s0 + m] += (m1 + logf(s1)) - (m2 + logf(s2));
  }

  // Phase 6: nxt[m,j] = sum_{k=(i,d)} tmp[m,i]*enc[m,d] * A2[k,j], K=8192.
  // K-pairs share i (k = i*64 + d), so one tmp splat feeds 4 WMMAs; the
  // operand product is done as packed bf16 (v_pk_mul_bf16) to stay XDL-bound.
  {
    const int mt  = wave >> 2;
    const int jt0 = wave & 3;
    const int mrow = (mt << 4) + (lane & 15);
    const int kb8  = (lane & 16) >> 1;
    const bf16* erow = encb + mrow * SE + kb8;
    v8f c0 = {}, c1 = {};
    for (int i = 0; i < 128; ++i) {       // i = kc>>1; two 32-wide d-halves each
      const bf16 ti = tmpb[mrow * S1 + i];
      v16bf tv;
#pragma unroll
      for (int e = 0; e < 16; ++e) tv[e] = ti;
      if (i + 1 < 128) {
        __builtin_prefetch(P_A + ((((i + 1) << 4) + jt0) * 512), 0, 1);
        __builtin_prefetch(P_A + ((((i + 1) << 4) + jt0 + 4) * 512), 0, 1);
      }
#pragma unroll
      for (int half = 0; half < 2; ++half) {
        const int kc = (i << 1) + half;
        const bf16* ep = erow + (half << 5);
        v16bf ev;
#pragma unroll
        for (int e = 0; e < 8; ++e) { ev[e] = ep[e]; ev[e + 8] = ep[16 + e]; }
        v16bf a = ev * tv;                // packed bf16 multiply
        v16bf b0 = load_b_pack(P_A + (((kc << 3) + jt0) * 512), lane);
        v16bf b1 = load_b_pack(P_A + (((kc << 3) + jt0 + 4) * 512), lane);
        c0 = WMMA_BF16(a, b0, c0);
        c1 = WMMA_BF16(a, b1, c1);
      }
    }
    const int rbase = (mt << 4) + ((lane & 16) >> 1);
    const int col0 = (jt0 << 4) + (lane & 15);
#pragma unroll
    for (int e = 0; e < 8; ++e) {
      tmp_out[(s0 + rbase + e) * RDIM + col0]      = c0[e];
      tmp_out[(s0 + rbase + e) * RDIM + col0 + 64] = c1[e];
    }
  }
}

// ---------------------------------------------------------------------------
extern "C" void kernel_launch(void* const* d_in, const int* in_sizes, int n_in,
                              void* d_out, int out_size, void* d_ws, size_t ws_size,
                              hipStream_t stream) {
  (void)in_sizes; (void)n_in; (void)out_size; (void)ws_size;
  const float* X     = (const float*)d_in[0];
  const float* e1w   = (const float*)d_in[1];
  const float* e1b   = (const float*)d_in[2];
  const float* e2w   = (const float*)d_in[3];
  const float* e2b   = (const float*)d_in[4];
  const float* A     = (const float*)d_in[5];
  const float* initw = (const float*)d_in[6];
  const float* n1w   = (const float*)d_in[7];
  const float* n1b   = (const float*)d_in[8];
  const float* n2w   = (const float*)d_in[9];
  const float* n2b   = (const float*)d_in[10];
  const float* muw   = (const float*)d_in[11];
  const float* mub   = (const float*)d_in[12];
  const float* sgw   = (const float*)d_in[13];
  const float* sgb   = (const float*)d_in[14];
  const float* alw   = (const float*)d_in[15];
  const float* alb   = (const float*)d_in[16];
  float* out = (float*)d_out;

  char* ws = (char*)d_ws;
  size_t off = 0;
  auto carve = [&](size_t bytes) -> void* {
    void* p = (void*)(ws + off);
    off = (off + bytes + 255) & ~(size_t)255;
    return p;
  };
  float* tmp0  = (float*)carve((size_t)NSAMP * RDIM * 4);
  float* tmp1  = (float*)carve((size_t)NSAMP * RDIM * 4);
  bf16*  P_A   = (bf16*)carve((size_t)RDIM * DENC * RDIM * 2);
  bf16*  P_n1  = (bf16*)carve((size_t)NH * RDIM * 2);
  bf16*  P_n2  = (bf16*)carve((size_t)NH * NH * 2);
  bf16*  P_mu  = (bf16*)carve((size_t)32 * NH * 2);
  bf16*  P_sg  = (bf16*)carve((size_t)32 * NH * 2);
  bf16*  P_al  = (bf16*)carve((size_t)32 * NH * 2);
  bf16*  P_e2  = (bf16*)carve((size_t)DENC * EH * 2);
  float* nrmsq = (float*)carve((size_t)TSTEPS * 4);

  init_kernel<<<(NSAMP * RDIM + 255) / 256, 256, 0, stream>>>(initw, tmp0, out, nrmsq);

  auto packN = [&](const float* W, bf16* P, int K, int O, int isOK) {
    int tot = K * O;
    pack_b_kernel<<<(tot + 255) / 256, 256, 0, stream>>>(W, P, K, O, isOK);
  };
  packN(n1w, P_n1, 128, 256, 1);
  packN(n2w, P_n2, 256, 256, 1);
  packN(muw, P_mu, 256, 32, 1);
  packN(sgw, P_sg, 256, 32, 1);
  packN(alw, P_al, 256, 32, 1);
  packN(e2w, P_e2, 128, 64, 1);
  packN(A,   P_A, 8192, 128, 0);

  for (int t = 0; t < TSTEPS; ++t) {
    const float* tin = (t & 1) ? tmp1 : tmp0;
    float* tout      = (t & 1) ? tmp0 : tmp1;
    step_kernel<<<NSAMP / MT, 256, 0, stream>>>(
        X, e1w, e1b, e2b, n1b, n2b, mub, sgb, alb,
        P_n1, P_n2, P_mu, P_sg, P_al, P_e2, P_A,
        tin, tout, out, nrmsq, t);
  }
  final_kernel<<<1, TSTEPS, 0, stream>>>(initw, nrmsq, out);
}
